// GATLayer_84731114815917
// MI455X (gfx1250) — matrix operand order repaired and checked by reading, not compiled, not run on hardware
//
#include <hip/hip_runtime.h>

#define N_NODES 50000
#define N_PAD   50048   // 391 blocks * 128 rows
#define N_EDGES 800000
#define F_INDIM 64
#define HC      128     // H*C
#define NHEAD   4
#define CDIM    32
#define NEG_SLOPE 0.2f
#define LN_EPS    1e-5f

#define XPITCH 72       // padded LDS pitch (halves): 144B = 9*16B -> b128-aligned, spread banks
#define BPITCH 72

typedef _Float16 v16h __attribute__((ext_vector_type(16)));
typedef _Float16 h8   __attribute__((ext_vector_type(8)));
typedef _Float16 h4   __attribute__((ext_vector_type(4)));
typedef float    v8f  __attribute__((ext_vector_type(8)));

__device__ __forceinline__ void atomAddF(float* p, float v) {
    __hip_atomic_fetch_add(p, v, __ATOMIC_RELAXED, __HIP_MEMORY_SCOPE_AGENT);
}

__device__ __forceinline__ float waveRedSum(float v) {
#pragma unroll
    for (int off = 16; off > 0; off >>= 1) v += __shfl_xor(v, off, 32);
    return v;
}

__device__ __forceinline__ v16h cat8(h8 lo, h8 hi) {
    return __builtin_shufflevector(lo, hi, 0, 1, 2, 3, 4, 5, 6, 7,
                                   8, 9, 10, 11, 12, 13, 14, 15);
}

// ---------------------------------------------------------------------------
// K1: fused GEMM  [h | resid] = x @ [W | res_w]  (+res_b on resid half)
// Block = 256 threads (8 waves); block does 128 rows; wave w -> 16 rows.
// f16 WMMA 16x16x32, f32 accumulate. resid lands in d_out (used as scratch).
// h region is padded to N_PAD rows -> unconditional stores; resid OOB lanes
// store to a dump row inside S (rewritten by k_node_init before any read).
// ---------------------------------------------------------------------------
__global__ __launch_bounds__(256) void k_gemm(
    const float* __restrict__ x, const float* __restrict__ W,
    const float* __restrict__ resW, const float* __restrict__ resB,
    float* __restrict__ h, float* __restrict__ resid, float* __restrict__ dump)
{
    __shared__ _Float16 sX[128 * XPITCH];   // 18 KB
    __shared__ _Float16 sBt[256 * BPITCH];  // 36 KB, transposed: sBt[col][k]
    const int t  = threadIdx.x;
    const int r0 = blockIdx.x * 128;

    // stage x tile: coalesced float4 reads, packed h4 LDS writes
    for (int i4 = t; i4 < 128 * 16; i4 += 256) {
        int row = i4 >> 4, c4 = (i4 & 15) * 4;
        int gr = r0 + row;
        float4 v = {0.f, 0.f, 0.f, 0.f};
        if (gr < N_NODES) v = *(const float4*)(x + gr * F_INDIM + c4);
        h4 hv = {(_Float16)v.x, (_Float16)v.y, (_Float16)v.z, (_Float16)v.w};
        *(h4*)&sX[row * XPITCH + c4] = hv;
    }
    // stage B = [W | res_w] transposed: coalesced float4 reads, strided b16 writes
    for (int i4 = t; i4 < 64 * 64; i4 += 256) {
        int row = i4 >> 6, c4 = (i4 & 63) * 4;
        const float* sp = (c4 < HC) ? (W + row * HC + c4)
                                    : (resW + row * HC + (c4 - HC));
        float4 v = *(const float4*)sp;
        sBt[(c4 + 0) * BPITCH + row] = (_Float16)v.x;
        sBt[(c4 + 1) * BPITCH + row] = (_Float16)v.y;
        sBt[(c4 + 2) * BPITCH + row] = (_Float16)v.z;
        sBt[(c4 + 3) * BPITCH + row] = (_Float16)v.w;
    }
    __syncthreads();

    const int wave = t >> 5, lane = t & 31;
    const int m0      = wave * 16;
    const int laneLo  = lane & 15;
    const int laneHiA = (lane & 16) ? 8 : 0;    // A frag K offset / C frag M offset
    const int laneHiB = (lane & 16) ? 16 : 0;   // unused (B frag K offset is laneHiA)

    // A fragments, both 32-wide K tiles: 2x ds_load_b128 each + concat
    const _Float16* xrow = &sX[(m0 + laneLo) * XPITCH];
    v16h a0 = cat8(*(const h8*)(xrow + laneHiA),      *(const h8*)(xrow + 16 + laneHiA));
    v16h a1 = cat8(*(const h8*)(xrow + 32 + laneHiA), *(const h8*)(xrow + 48 + laneHiA));

#pragma unroll 1
    for (int jt = 0; jt < 16; ++jt) {
        const int col16 = jt * 16 + laneLo;
        const _Float16* bp = &sBt[col16 * BPITCH + laneHiA];
        v16h b0 = cat8(*(const h8*)(bp),      *(const h8*)(bp + 16));
        v16h b1 = cat8(*(const h8*)(bp + 32), *(const h8*)(bp + 48));

        v8f acc = {};
        acc = __builtin_amdgcn_wmma_f32_16x16x32_f16(false, a0, false, b0,
                                                     (short)0, acc, false, false);
        acc = __builtin_amdgcn_wmma_f32_16x16x32_f16(false, a1, false, b1,
                                                     (short)0, acc, false, false);

        // C layout: VGPR r -> M=r (lanes 0-15) / M=r+8 (lanes 16-31), N=lane&15
        const int grBase = r0 + m0 + laneHiA;
        if (jt < 8) {                 // uniform branch: h half, padded -> no guard
#pragma unroll
            for (int r = 0; r < 8; ++r)
                h[(size_t)(grBase + r) * HC + col16] = acc[r];
        } else {                      // resid half -> d_out, branchless addr select
            const int rc = col16 - HC;
            const float rb = resB[rc];
#pragma unroll
            for (int r = 0; r < 8; ++r) {
                int gr = grBase + r;
                float* p = (gr < N_NODES) ? (resid + (size_t)gr * HC + rc)
                                          : (dump + rc);
                *p = acc[r] + rb;
            }
        }
    }
}

// ---------------------------------------------------------------------------
// K2: per-node attention logits + self-loop init of denom and S.
// One wave per node; 4 channels per lane.
// ---------------------------------------------------------------------------
__global__ __launch_bounds__(256) void k_node_init(
    const float* __restrict__ h, const float* __restrict__ attS,
    const float* __restrict__ attD,
    float* __restrict__ aS, float* __restrict__ aD,
    float* __restrict__ denom, float* __restrict__ S)
{
    const int wave = threadIdx.x >> 5, lane = threadIdx.x & 31;
    const int n = blockIdx.x * 8 + wave;
    if (n >= N_NODES) return;

    float hv[NHEAD], as[NHEAD], ad[NHEAD];
#pragma unroll
    for (int i = 0; i < NHEAD; ++i) {
        hv[i] = h[(size_t)n * HC + i * CDIM + lane];
        as[i] = waveRedSum(hv[i] * attS[i * CDIM + lane]);
        ad[i] = waveRedSum(hv[i] * attD[i * CDIM + lane]);
    }
#pragma unroll
    for (int i = 0; i < NHEAD; ++i) {
        float e = as[i] + ad[i];                 // self loop: src == dst == n
        e = (e >= 0.0f) ? e : NEG_SLOPE * e;
        float ex = __expf(e);                    // logits O(1): no max-shift needed
        S[(size_t)n * HC + i * CDIM + lane] = hv[i] * ex;
        if (lane == 0) {
            aS[n * NHEAD + i]    = as[i];
            aD[n * NHEAD + i]    = ad[i];
            denom[n * NHEAD + i] = ex;
        }
    }
}

// ---------------------------------------------------------------------------
// K3: single edge pass. One wave per edge.
//   ex[h] = exp(leaky_relu(a_src[s]+a_dst[d]))
//   denom[d][h] += ex[h];  S[d][:] += ex[head(c)] * h[s][:]
// Accumulators (~26 MB) are L2-resident on MI455X (192 MB L2).
// ---------------------------------------------------------------------------
__global__ __launch_bounds__(256) void k_edge(
    const long long* __restrict__ ei,
    const float* __restrict__ h,
    const float* __restrict__ aS, const float* __restrict__ aD,
    float* __restrict__ denom, float* __restrict__ S)
{
    const int wid  = (int)((blockIdx.x * 256u + threadIdx.x) >> 5);
    const int lane = threadIdx.x & 31;
    if (wid >= N_EDGES) return;

    const int src = (int)ei[wid];
    const int dst = (int)ei[N_EDGES + wid];

    const float4 a4 = *(const float4*)(aS + src * 4);
    const float4 b4 = *(const float4*)(aD + dst * 4);
    float ex[NHEAD];
    {
        float e0 = a4.x + b4.x, e1 = a4.y + b4.y, e2 = a4.z + b4.z, e3 = a4.w + b4.w;
        e0 = (e0 >= 0.f) ? e0 : NEG_SLOPE * e0;
        e1 = (e1 >= 0.f) ? e1 : NEG_SLOPE * e1;
        e2 = (e2 >= 0.f) ? e2 : NEG_SLOPE * e2;
        e3 = (e3 >= 0.f) ? e3 : NEG_SLOPE * e3;
        ex[0] = __expf(e0); ex[1] = __expf(e1); ex[2] = __expf(e2); ex[3] = __expf(e3);
    }
    if (lane < NHEAD) {
        float v = (lane == 0) ? ex[0] : (lane == 1) ? ex[1] : (lane == 2) ? ex[2] : ex[3];
        atomAddF(&denom[dst * NHEAD + lane], v);
    }
#pragma unroll
    for (int i = 0; i < NHEAD; ++i) {
        float hv = h[(size_t)src * HC + i * CDIM + lane];   // coalesced 128B gather
        atomAddF(&S[(size_t)dst * HC + i * CDIM + lane], hv * ex[i]);
    }
}

// ---------------------------------------------------------------------------
// K4: finalize per node: out = LN(elu(S/denom + bias) + resid)
// resid was stashed in d_out by k_gemm; overwrite in place.
// ---------------------------------------------------------------------------
__global__ __launch_bounds__(256) void k_final(
    const float* __restrict__ S, const float* __restrict__ denom,
    const float* __restrict__ bias, const float* __restrict__ lng,
    const float* __restrict__ lnb, float* __restrict__ out)
{
    const int wave = threadIdx.x >> 5, lane = threadIdx.x & 31;
    const int n = blockIdx.x * 8 + wave;
    if (n >= N_NODES) return;

    const float4 dn = *(const float4*)(denom + n * NHEAD);
    const float dnv[NHEAD] = {dn.x, dn.y, dn.z, dn.w};
    float v[NHEAD];
    float s = 0.f;
#pragma unroll
    for (int i = 0; i < NHEAD; ++i) {
        int ch = i * CDIM + lane;
        float tv = S[(size_t)n * HC + ch] / dnv[i] + bias[ch];
        tv = (tv > 0.f) ? tv : expm1f(tv);       // elu
        tv += out[(size_t)n * HC + ch];          // residual (+res_b already applied)
        v[i] = tv; s += tv;
    }
    const float mu = waveRedSum(s) * (1.0f / HC);
    float var = 0.f;
#pragma unroll
    for (int i = 0; i < NHEAD; ++i) { float d = v[i] - mu; var += d * d; }
    var = waveRedSum(var) * (1.0f / HC);
    const float rs = rsqrtf(var + LN_EPS);
#pragma unroll
    for (int i = 0; i < NHEAD; ++i) {
        int ch = i * CDIM + lane;
        out[(size_t)n * HC + ch] = lng[ch] * (v[i] - mu) * rs + lnb[ch];
    }
}

// ---------------------------------------------------------------------------
extern "C" void kernel_launch(void* const* d_in, const int* in_sizes, int n_in,
                              void* d_out, int out_size, void* d_ws, size_t ws_size,
                              hipStream_t stream) {
    const float*     x    = (const float*)d_in[0];
    const long long* ei   = (const long long*)d_in[1];
    const float*     W    = (const float*)d_in[2];
    const float*     attS = (const float*)d_in[3];
    const float*     attD = (const float*)d_in[4];
    const float*     bias = (const float*)d_in[5];
    const float*     resW = (const float*)d_in[6];
    const float*     resB = (const float*)d_in[7];
    const float*     lng  = (const float*)d_in[8];
    const float*     lnb  = (const float*)d_in[9];
    float* out = (float*)d_out;

    // ws layout (floats): h[N_PAD*128] | S[N*128] | aS[N*4] | aD[N*4] | denom[N*4]
    float* ws    = (float*)d_ws;
    float* h     = ws;
    float* S     = h  + (size_t)N_PAD * HC;
    float* aS    = S  + (size_t)N_NODES * HC;
    float* aD    = aS + (size_t)N_NODES * NHEAD;
    float* denom = aD + (size_t)N_NODES * NHEAD;

    dim3 blk(256);
    // k_gemm dumps OOB resid lanes into S[0:128] -- rewritten by k_node_init.
    k_gemm<<<(N_NODES + 127) / 128, blk, 0, stream>>>(x, W, resW, resB, h, out, S);
    k_node_init<<<(N_NODES + 7) / 8, blk, 0, stream>>>(h, attS, attD, aS, aD, denom, S);
    k_edge<<<(N_EDGES + 7) / 8, blk, 0, stream>>>(ei, h, aS, aD, denom, S);
    k_final<<<(N_NODES + 7) / 8, blk, 0, stream>>>(S, denom, bias, lng, lnb, out);
}
